// Transformer_55791625175358
// MI455X (gfx1250) — compile-verified, matrix operand order checked by reading, not run on hardware
//
#include <hip/hip_runtime.h>
#include <stdint.h>

#define CTX 2048
#define DM  768
#define DM3 2304
#define NH  12
#define HD  64
#define NL  4
#define NV  50257
#define NVP 50304
#define FF  3072

typedef __attribute__((ext_vector_type(16))) __bf16 v16bf;
typedef __attribute__((ext_vector_type(8)))  __bf16 v8bf;
typedef __attribute__((ext_vector_type(8)))  float  v8f;

// ---------- helpers ----------

__device__ __forceinline__ uint16_t f2bf(float f) {
  union { float f; uint32_t u; } c; c.f = f;
  uint32_t u = c.u;
  if ((u & 0x7F800000u) != 0x7F800000u)           // not inf/nan: RNE round
    u = u + 0x7FFFu + ((u >> 16) & 1u);
  return (uint16_t)(u >> 16);
}

__device__ __forceinline__ float gelu_f(float x) {
  const float c = 0.7978845608028654f;            // sqrt(2/pi)
  float t = tanhf(c * (x + 0.044715f * x * x * x));
  return 0.5f * x * (1.0f + t);
}

// Load one 16x32 bf16 WMMA A/B fragment for this lane.
// run32 points at the 32-element K-run for this lane's row (A) / column (B).
// Element e<8  -> K = half*8 + e ; element e>=8 -> K = 16 + half*8 + (e-8).
__device__ __forceinline__ v16bf load_frag(const uint16_t* run32, int half) {
  const v8bf* p0 = (const v8bf*)(run32 + half * 8);
  const v8bf* p1 = (const v8bf*)(run32 + 16 + half * 8);
  v8bf lo = *p0, hi = *p1;
  v16bf r;
#pragma unroll
  for (int e = 0; e < 8; ++e) { r[e] = lo[e]; r[8 + e] = hi[e]; }
  return r;
}

// ---------- elementwise / prep kernels ----------

__global__ void k_embed(const int* __restrict__ tok, const float* __restrict__ vemb,
                        const float* __restrict__ pemb, float* __restrict__ x) {
  int i = blockIdx.x * blockDim.x + threadIdx.x;
  if (i >= CTX * DM) return;
  int c = i / DM, d = i % DM;
  x[i] = vemb[(size_t)tok[c] * DM + d] + pemb[i];
}

// one block (256 thr = 8 waves) per row
__global__ void k_layernorm(const float* __restrict__ x, const float* __restrict__ g,
                            const float* __restrict__ b, float* __restrict__ yf,
                            uint16_t* __restrict__ ybf) {
  int row = blockIdx.x;
  const float* xr = x + (size_t)row * DM;
  __shared__ float red[8];
  int lane = threadIdx.x & 31, wid = threadIdx.x >> 5;

  float s = 0.f;
  for (int d = threadIdx.x; d < DM; d += 256) s += xr[d];
#pragma unroll
  for (int off = 16; off >= 1; off >>= 1) s += __shfl_xor(s, off, 32);
  if (lane == 0) red[wid] = s;
  __syncthreads();
  float tot = 0.f;
#pragma unroll
  for (int i = 0; i < 8; ++i) tot += red[i];
  float mu = tot / DM;
  __syncthreads();

  float vs = 0.f;
  for (int d = threadIdx.x; d < DM; d += 256) { float t = xr[d] - mu; vs += t * t; }
#pragma unroll
  for (int off = 16; off >= 1; off >>= 1) vs += __shfl_xor(vs, off, 32);
  if (lane == 0) red[wid] = vs;
  __syncthreads();
  float vt = 0.f;
#pragma unroll
  for (int i = 0; i < 8; ++i) vt += red[i];
  float inv = rsqrtf(vt / DM + 1e-5f);

  for (int d = threadIdx.x; d < DM; d += 256) {
    float y = (xr[d] - mu) * inv * g[d] + b[d];
    yf[(size_t)row * DM + d] = y;
    ybf[(size_t)row * DM + d] = f2bf(y);
  }
}

// W[K][N] fp32 -> Wt[Npad][K] bf16 (zero pad rows n >= N).
// 32x32 tile transpose through LDS: coalesced reads along N, coalesced writes along K.
__global__ void k_wconv(const float* __restrict__ w, uint16_t* __restrict__ wt,
                        int K, int N, int Npad) {
  __shared__ uint16_t tile[32][33];
  int tk = blockIdx.x * 32;                 // K tile origin
  int tn = blockIdx.y * 32;                 // N tile origin
  int tx = threadIdx.x & 31;
  int ty = threadIdx.x >> 5;                // 0..7
#pragma unroll
  for (int i = 0; i < 4; ++i) {
    int k = tk + ty + 8 * i, n = tn + tx;
    float v = (k < K && n < N) ? w[(size_t)k * N + n] : 0.f;
    tile[tx][ty + 8 * i] = f2bf(v);         // tile[n_local][k_local]
  }
  __syncthreads();
#pragma unroll
  for (int i = 0; i < 4; ++i) {
    int n = tn + ty + 8 * i, k = tk + tx;
    if (n < Npad && k < K) wt[(size_t)n * K + k] = tile[ty + 8 * i][tx];
  }
}

// v slice of qkv (bf16) -> vT[d][c] bf16  (d in 0..DM-1 across all heads)
__global__ void k_vtrans(const uint16_t* __restrict__ qkv, uint16_t* __restrict__ vt) {
  int i = blockIdx.x * blockDim.x + threadIdx.x;
  if (i >= DM * CTX) return;
  int d = i / CTX, c = i % CTX;
  vt[i] = qkv[(size_t)c * DM3 + 2 * DM + d];
}

// ---------- WMMA GEMM: out[M][N] = A[M][K](bf16) x Bt[N][K](bf16)^T + bias (+res)(gelu) ----------
// one wave computes a 32(M) x 64(N) tile: 2 A fragments, 8 accumulators,
// 12 b128 loads per 8 WMMAs per K-step of 32.
__global__ void k_gemm(const uint16_t* __restrict__ A, const uint16_t* __restrict__ Bt,
                       const float* __restrict__ bias, const float* __restrict__ res,
                       float* __restrict__ outf, uint16_t* __restrict__ outb,
                       int M, int K, int N, int Npad, int gelu) {
  int ntiles = Npad / 64;
  int wave = blockIdx.x * (blockDim.x >> 5) + (threadIdx.x >> 5);
  if (wave >= (M / 32) * ntiles) return;            // wave-uniform: EXEC stays full
  int mt = wave / ntiles, nt = wave % ntiles;
  int m0 = mt * 32, n0 = nt * 64;
  int lane = threadIdx.x & 31;
  int r = lane & 15, half = lane >> 4;

  v8f acc[2][4] = {};
  const uint16_t* arow0 = A + (size_t)(m0 + r) * K;
  const uint16_t* arow1 = A + (size_t)(m0 + 16 + r) * K;

  for (int k = 0; k < K; k += 32) {
    __builtin_prefetch(arow0 + k + 128, 0, 1);
    v16bf a0 = load_frag(arow0 + k, half);
    v16bf a1 = load_frag(arow1 + k, half);
#pragma unroll
    for (int s = 0; s < 4; ++s) {
      const uint16_t* bcol = Bt + (size_t)(n0 + s * 16 + r) * K;
      v16bf b = load_frag(bcol + k, half);
      acc[0][s] = __builtin_amdgcn_wmma_f32_16x16x32_bf16(
          false, a0, false, b, (short)0, acc[0][s], false, false);
      acc[1][s] = __builtin_amdgcn_wmma_f32_16x16x32_bf16(
          false, a1, false, b, (short)0, acc[1][s], false, false);
    }
  }

#pragma unroll
  for (int s = 0; s < 4; ++s) {
    int n = n0 + s * 16 + r;
    if (n >= N) continue;
    float bv = bias ? bias[n] : 0.f;
#pragma unroll
    for (int g2 = 0; g2 < 2; ++g2) {
#pragma unroll
      for (int v = 0; v < 8; ++v) {
        int m = m0 + g2 * 16 + v + half * 8;
        float val = acc[g2][s][v] + bv;
        if (gelu) val = gelu_f(val);
        if (res)  val += res[(size_t)m * N + n];
        if (outf) outf[(size_t)m * N + n] = val;
        if (outb) outb[(size_t)m * N + n] = f2bf(val);
      }
    }
  }
}

// ---------- flash attention: one wave per (head, 16-query tile) ----------
__global__ void k_attn(const uint16_t* __restrict__ qkv, const uint16_t* __restrict__ vT,
                       uint16_t* __restrict__ o) {
  const int QT = CTX / 16;
  int wave = blockIdx.x * 8 + (threadIdx.x >> 5);
  if (wave >= NH * QT) return;
  int h = wave / QT, qt = wave % QT;
  int q0 = qt * 16;
  int lane = threadIdx.x & 31;
  int r = lane & 15, half = lane >> 4;

  __shared__ __align__(16) uint16_t pshare[8][16 * 32];   // per-wave P staging (bf16)
  uint16_t* pw = &pshare[threadIdx.x >> 5][0];

  // Q fragments (16 rows x 64 head-dims -> two K=32 fragments), loaded once
  v16bf aq[2];
  {
    const uint16_t* qrow = qkv + (size_t)(q0 + r) * DM3 + h * HD;
#pragma unroll
    for (int c2 = 0; c2 < 2; ++c2) aq[c2] = load_frag(qrow + c2 * 32, half);
  }

  float mrun[8], lrun[8];
#pragma unroll
  for (int v = 0; v < 8; ++v) { mrun[v] = -3.0e38f; lrun[v] = 0.f; }
  v8f oacc[4] = {};

  int kend = q0 + 16;                                // exclusive key bound (causal)
  for (int kb = 0; kb < kend; kb += 32) {
    // ---- scores S = Q x K^T for 32 keys ----
    v8f s[2] = {};
#pragma unroll
    for (int sub = 0; sub < 2; ++sub) {
      const uint16_t* krow = qkv + (size_t)(kb + sub * 16 + r) * DM3 + DM + h * HD;
#pragma unroll
      for (int c2 = 0; c2 < 2; ++c2) {
        v16bf b = load_frag(krow + c2 * 32, half);
        s[sub] = __builtin_amdgcn_wmma_f32_16x16x32_bf16(
            false, aq[c2], false, b, (short)0, s[sub], false, false);
      }
    }

    // ---- scale, causal mask, online softmax ----
    float mnew[8], alpha[8];
#pragma unroll
    for (int v = 0; v < 8; ++v) {
      int row = q0 + v + half * 8;
#pragma unroll
      for (int sub = 0; sub < 2; ++sub) {
        int col = kb + sub * 16 + r;
        float sv = s[sub][v] * 0.125f;               // 1/sqrt(64)
        s[sub][v] = (col <= row) ? sv : -3.0e38f;
      }
      float bm = fmaxf(s[0][v], s[1][v]);
#pragma unroll
      for (int off = 1; off < 16; off <<= 1) bm = fmaxf(bm, __shfl_xor(bm, off, 32));
      mnew[v] = fmaxf(mrun[v], bm);
      alpha[v] = __expf(mrun[v] - mnew[v]);
    }
#pragma unroll
    for (int v = 0; v < 8; ++v) {
      float ps = 0.f;
#pragma unroll
      for (int sub = 0; sub < 2; ++sub) {
        float p = __expf(s[sub][v] - mnew[v]);
        s[sub][v] = p;
        ps += p;
      }
#pragma unroll
      for (int off = 1; off < 16; off <<= 1) ps += __shfl_xor(ps, off, 32);
      lrun[v] = lrun[v] * alpha[v] + ps;
      mrun[v] = mnew[v];
    }

    // ---- transpose P from C-layout to A-layout via per-wave LDS ----
#pragma unroll
    for (int v = 0; v < 8; ++v) {
      int pr = v + half * 8;
#pragma unroll
      for (int sub = 0; sub < 2; ++sub)
        pw[pr * 32 + sub * 16 + r] = f2bf(s[sub][v]);
    }
    asm volatile("s_wait_dscnt 0" ::: "memory");
    v16bf ap = load_frag(pw + r * 32, half);

    // ---- O = alpha*O + P x V ----
#pragma unroll
    for (int n = 0; n < 4; ++n) {
#pragma unroll
      for (int v = 0; v < 8; ++v) oacc[n][v] *= alpha[v];
      const uint16_t* vrow = vT + (size_t)(h * HD + n * 16 + r) * CTX + kb;
      v16bf b = load_frag(vrow, half);
      oacc[n] = __builtin_amdgcn_wmma_f32_16x16x32_bf16(
          false, ap, false, b, (short)0, oacc[n], false, false);
    }
  }

  // ---- normalize and store O (bf16) ----
#pragma unroll
  for (int n = 0; n < 4; ++n) {
#pragma unroll
    for (int v = 0; v < 8; ++v) {
      int m = q0 + v + half * 8;
      int d = h * HD + n * 16 + r;
      o[(size_t)m * DM + d] = f2bf(oacc[n][v] / lrun[v]);
    }
  }
}

// ---------- host ----------

static void launch_gemm(const uint16_t* A, const uint16_t* Bt, const float* bias,
                        const float* res, float* outf, uint16_t* outb,
                        int M, int K, int N, int Npad, int gelu, hipStream_t stream) {
  int waves = (M / 32) * (Npad / 64);
  int blocks = (waves + 7) / 8;
  k_gemm<<<blocks, 256, 0, stream>>>(A, Bt, bias, res, outf, outb, M, K, N, Npad, gelu);
}

static void launch_wconv(const float* w, uint16_t* wt, int K, int N, int Npad,
                         hipStream_t stream) {
  dim3 grid((K + 31) / 32, (Npad + 31) / 32);
  k_wconv<<<grid, 256, 0, stream>>>(w, wt, K, N, Npad);
}

extern "C" void kernel_launch(void* const* d_in, const int* in_sizes, int n_in,
                              void* d_out, int out_size, void* d_ws, size_t ws_size,
                              hipStream_t stream) {
  (void)in_sizes; (void)n_in; (void)out_size; (void)ws_size;
  const int*   tokens = (const int*)  d_in[0];
  const float* vemb   = (const float*)d_in[1];
  const float* pemb   = (const float*)d_in[2];
  const float* ln1_g  = (const float*)d_in[3];
  const float* ln1_b  = (const float*)d_in[4];
  const float* att_w  = (const float*)d_in[5];
  const float* att_b  = (const float*)d_in[6];
  const float* apw    = (const float*)d_in[7];
  const float* apb    = (const float*)d_in[8];
  const float* ln2_g  = (const float*)d_in[9];
  const float* ln2_b  = (const float*)d_in[10];
  const float* fc_w   = (const float*)d_in[11];
  const float* fc_b   = (const float*)d_in[12];
  const float* mpw    = (const float*)d_in[13];
  const float* mpb    = (const float*)d_in[14];
  const float* lnf_g  = (const float*)d_in[15];
  const float* lnf_b  = (const float*)d_in[16];
  const float* out_w  = (const float*)d_in[17];
  const float* out_b  = (const float*)d_in[18];

  char* ws = (char*)d_ws;
  size_t off = 0;
  auto take = [&](size_t bytes) -> void* {
    void* p = ws + off;
    off = (off + bytes + 255) & ~(size_t)255;
    return p;
  };
  float*    xf   = (float*)   take((size_t)CTX * DM * 4);
  float*    hres = (float*)   take((size_t)CTX * DM * 4);
  uint16_t* hbf  = (uint16_t*)take((size_t)CTX * DM * 2);
  uint16_t* qkv  = (uint16_t*)take((size_t)CTX * DM3 * 2);
  uint16_t* vT   = (uint16_t*)take((size_t)DM * CTX * 2);
  uint16_t* obf  = (uint16_t*)take((size_t)CTX * DM * 2);
  uint16_t* mlp  = (uint16_t*)take((size_t)CTX * FF * 2);
  uint16_t* wt   = (uint16_t*)take((size_t)NVP * DM * 2);

  k_embed<<<(CTX * DM + 255) / 256, 256, 0, stream>>>(tokens, vemb, pemb, xf);

  for (int l = 0; l < NL; ++l) {
    // h = LN1(x)
    k_layernorm<<<CTX, 256, 0, stream>>>(xf, ln1_g + l * DM, ln1_b + l * DM, hres, hbf);
    // qkv = h @ att_w + att_b
    launch_wconv(att_w + (size_t)l * DM * DM3, wt, DM, DM3, DM3, stream);
    launch_gemm(hbf, wt, att_b + l * DM3, nullptr, nullptr, qkv, CTX, DM, DM3, DM3, 0, stream);
    // attention
    k_vtrans<<<(DM * CTX + 255) / 256, 256, 0, stream>>>(qkv, vT);
    k_attn<<<(NH * (CTX / 16)) / 8, 256, 0, stream>>>(qkv, vT, obf);
    // x = h + o @ attn_proj_w + b
    launch_wconv(apw + (size_t)l * DM * DM, wt, DM, DM, DM, stream);
    launch_gemm(obf, wt, apb + l * DM, hres, xf, nullptr, CTX, DM, DM, DM, 0, stream);
    // h2 = LN2(x)
    k_layernorm<<<CTX, 256, 0, stream>>>(xf, ln2_g + l * DM, ln2_b + l * DM, hres, hbf);
    // mlp = gelu(h2 @ fc_w + fc_b)
    launch_wconv(fc_w + (size_t)l * DM * FF, wt, DM, FF, FF, stream);
    launch_gemm(hbf, wt, fc_b + l * FF, nullptr, nullptr, mlp, CTX, DM, FF, FF, 1, stream);
    // x = h2 + mlp @ mlp_proj_w + b
    launch_wconv(mpw + (size_t)l * FF * DM, wt, FF, DM, DM, stream);
    launch_gemm(mlp, wt, mpb + l * DM, hres, xf, nullptr, CTX, FF, DM, DM, 0, stream);
  }

  // final LN + logits
  k_layernorm<<<CTX, 256, 0, stream>>>(xf, lnf_g, lnf_b, hres, hbf);
  launch_wconv(out_w, wt, DM, NV, NVP, stream);
  launch_gemm(hbf, wt, out_b, nullptr, (float*)d_out, nullptr, CTX, DM, NV, NVP, 0, stream);
}
